// pyg_GAT_10282151707719
// MI455X (gfx1250) — matrix operand order repaired and checked by reading, not compiled, not run on hardware
//
#include <hip/hip_runtime.h>

// ---------------- problem constants ----------------
#define NNODES 50000
#define NEDGES 800000
#define ETOT   (NEDGES + NNODES)   // 850000 edges incl. self loops

typedef __bf16 bf16;
typedef __attribute__((ext_vector_type(16))) __bf16 v16bf;
typedef __attribute__((ext_vector_type(8)))  __bf16 v8bf;
typedef __attribute__((ext_vector_type(8)))  float  v8f;

// round-to-nearest-even f32 -> bf16
__device__ __forceinline__ bf16 f2bf(float f) {
    unsigned u = __float_as_uint(f);
    unsigned r = u + 0x7FFFu + ((u >> 16) & 1u);
    unsigned short s = (unsigned short)(r >> 16);
    return __builtin_bit_cast(bf16, s);
}

__device__ __forceinline__ void edge_nodes(const int* __restrict__ ei, int e, int& s, int& d) {
    if (e < NEDGES) { s = ei[e]; d = ei[NEDGES + e]; }  // edge_index[0]=src, [1]=dst
    else            { s = e - NEDGES; d = s; }          // self loop
}

__device__ __forceinline__ void atomicMaxF(float* addr, float v) {
    if (v >= 0.f) atomicMax((int*)addr, __float_as_int(v));
    else          atomicMin((unsigned int*)addr, __float_as_uint(v));
}

// ---------------- elementwise f32 -> bf16 ----------------
__global__ void f32_to_bf16_k(const float* __restrict__ in, bf16* __restrict__ out, int n) {
    int i = blockIdx.x * 256 + threadIdx.x;
    if (i < n) out[i] = f2bf(in[i]);
}

// ---------------- pack W[K,Ntrue] (f32) into WMMA B-fragment layout (bf16) ----
// Packed layout: [kt][nt][lane][16 bf16]. Lane<16 holds col n=lane, K 0..7 and 16..23
// of the 32-K tile; lane>=16 holds col n=lane-16, K 8..15 and 24..31.
__global__ void pack_w_k(const float* __restrict__ W, bf16* __restrict__ Bp,
                         int K, int Ntrue, int nTiles) {
    int idx = blockIdx.x * 256 + threadIdx.x;
    int total = (K >> 5) * nTiles * 512;
    if (idx >= total) return;
    int e    = idx & 15;
    int lane = (idx >> 4) & 31;
    int tile = idx >> 9;               // kt*nTiles + nt
    int nt = tile % nTiles;
    int kt = tile / nTiles;
    int v = e >> 1, p = e & 1;
    int kk = ((lane >> 4) ? 8 : 0) + (e < 8 ? e : e + 8);
    int k = kt * 32 + kk;
    int n = nt * 16 + (lane & 15);
    float val = (n < Ntrue) ? W[(size_t)k * Ntrue + n] : 0.f;
    Bp[idx] = f2bf(val);
}

// ---------------- bf16 WMMA GEMM: C[M,NTRUE] = A[M,K] @ W -------------------
// Fully specialized at compile time: KT k-tiles of 32, NT total n-tiles,
// JT n-tiles per wave (NT % JT == 0), NTRUE real column count.
// One wave computes a 16-row x (JT*16)-col tile; A fragment reused JT times.
// All WMMA control flow is wave-uniform / unrolled -> EXEC stays all-ones.
template<int KT, int NT, int JT, int NTRUE>
__global__ void gat_gemm_wmma(const bf16* __restrict__ A, const bf16* __restrict__ Bp,
                              float* __restrict__ C) {
    constexpr int K  = KT * 32;
    constexpr int NG = NT / JT;          // wave groups across columns
    const int wid  = (blockIdx.x * blockDim.x + threadIdx.x) >> 5;
    const int lane = threadIdx.x & 31;
    const int rowTile = wid / NG;
    const int group   = wid % NG;
    if (rowTile >= NNODES / 16) return;  // wave-uniform early out

    // A fragment addressing (16x32 bf16 layout, ISA 7.12.2):
    // lanes 0-15: row M=lane, VGPR0-3 = K 0..7, VGPR4-7 = K 16..23
    // lanes16-31: row M=lane-16, K 8..15 and 24..31
    const int m   = lane & 15;
    const int hiA = (lane >> 4) << 3;    // 0 or 8
    const bf16* arow = A + (size_t)(rowTile * 16 + m) * K;

    v8f zero = {0.f, 0.f, 0.f, 0.f, 0.f, 0.f, 0.f, 0.f};
    v8f acc[JT];
    #pragma unroll
    for (int j = 0; j < JT; ++j) acc[j] = zero;

    #pragma unroll
    for (int kt = 0; kt < KT; ++kt) {
        const int kb = kt << 5;
        v8bf alo = *reinterpret_cast<const v8bf*>(arow + kb + hiA);       // b128
        v8bf ahi = *reinterpret_cast<const v8bf*>(arow + kb + 16 + hiA);  // b128
        v16bf a;
        #pragma unroll
        for (int i = 0; i < 8; ++i) { a[i] = alo[i]; a[i + 8] = ahi[i]; }

        #pragma unroll
        for (int jj = 0; jj < JT; ++jj) {
            const int nt = group * JT + jj;
            const bf16* bl = Bp + (((size_t)(kt * NT + nt) * 32 + lane) << 4);
            v8bf blo = *reinterpret_cast<const v8bf*>(bl);      // b128
            v8bf bhi = *reinterpret_cast<const v8bf*>(bl + 8);  // b128
            v16bf b;
            #pragma unroll
            for (int i = 0; i < 8; ++i) { b[i] = blo[i]; b[i + 8] = bhi[i]; }
            acc[jj] = __builtin_amdgcn_wmma_f32_16x16x32_bf16(
                false, a, false, b, (short)0, acc[jj], false, false);
        }
    }

    // C/D layout: lane<16 -> col n=lane, rows v (VGPR0-7); lane>=16 -> rows v+8
    const int n     = lane & 15;
    const int rbase = rowTile * 16 + ((lane >> 4) << 3);
    #pragma unroll
    for (int jj = 0; jj < JT; ++jj) {
        const int col = (group * JT + jj) * 16 + n;
        if (col < NTRUE) {               // folds to true when NT*16 == NTRUE
            #pragma unroll
            for (int v = 0; v < 8; ++v)
                C[(size_t)(rbase + v) * NTRUE + col] = acc[jj][v];
        }
    }
}

// ---------------- per-node attention logits + softmax-state init ------------
__global__ void att_logits_k(const float* __restrict__ feat,
                             const float* __restrict__ a_s, const float* __restrict__ a_d,
                             float* __restrict__ als, float* __restrict__ ald,
                             float* __restrict__ emax, float* __restrict__ denom,
                             int H, int C) {
    int idx = blockIdx.x * 256 + threadIdx.x;
    if (idx >= NNODES * H) return;
    int nn = idx / H, h = idx - nn * H;
    const float* fp = feat + (size_t)nn * H * C + h * C;
    const float* as = a_s + h * C;
    const float* ad = a_d + h * C;
    float ss = 0.f, sd = 0.f;
    for (int i = 0; i < C; ++i) { float f = fp[i]; ss += f * as[i]; sd += f * ad[i]; }
    als[idx] = ss;
    ald[idx] = sd;
    emax[idx]  = -INFINITY;
    denom[idx] = 0.f;
}

// ---------------- edge score (LeakyReLU 0.2) + segment max ------------------
__global__ void edge_score_max_k(const float* __restrict__ als, const float* __restrict__ ald,
                                 const int* __restrict__ ei,
                                 float* __restrict__ esc, float* __restrict__ emax, int H) {
    int idx = blockIdx.x * 256 + threadIdx.x;
    if (idx >= ETOT * H) return;
    int e = idx / H, h = idx - e * H;
    int s, d; edge_nodes(ei, e, s, d);
    float sc = als[s * H + h] + ald[d * H + h];
    sc = (sc > 0.f) ? sc : 0.2f * sc;
    esc[idx] = sc;
    atomicMaxF(&emax[d * H + h], sc);
}

// ---------------- exp(e - emax) + segment sum -------------------------------
__global__ void edge_exp_sum_k(const float* __restrict__ esc, const float* __restrict__ emax,
                               const int* __restrict__ ei,
                               float* __restrict__ ew, float* __restrict__ denom, int H) {
    int idx = blockIdx.x * 256 + threadIdx.x;
    if (idx >= ETOT * H) return;
    int e = idx / H, h = idx - e * H;
    int s, d; edge_nodes(ei, e, s, d);
    float w = __expf(esc[idx] - emax[d * H + h]);
    ew[idx] = w;
    atomicAdd(&denom[d * H + h], w);
}

// ---------------- weighted scatter-add aggregation: one wave per edge -------
__global__ void aggregate_k(const float* __restrict__ feat, const float* __restrict__ ew,
                            const float* __restrict__ denom, const int* __restrict__ ei,
                            float* __restrict__ agg, int H, int C) {
    int wid  = (blockIdx.x * blockDim.x + threadIdx.x) >> 5;
    int lane = threadIdx.x & 31;
    if (wid >= ETOT) return;
    int s, d; edge_nodes(ei, wid, s, d);
    int hc = H * C;
    const float* fs = feat + (size_t)s * hc;
    float* od = agg + (size_t)d * hc;
    for (int i = lane; i < hc; i += 32) {
        int h = i / C;
        float alpha = ew[(size_t)wid * H + h] / denom[d * H + h];
        atomicAdd(&od[i], alpha * fs[i]);
    }
}

// ---------------- bias (+ELU) -> next-layer bf16 activation or fp32 output --
__global__ void finalize_k(const float* __restrict__ agg, const float* __restrict__ bias,
                           bf16* __restrict__ out_bf, float* __restrict__ out_f,
                           int total, int hc, int do_elu) {
    int i = blockIdx.x * 256 + threadIdx.x;
    if (i >= total) return;
    float v = agg[i] + bias[i % hc];
    if (do_elu) v = (v > 0.f) ? v : (__expf(v) - 1.f);
    if (out_bf) out_bf[i] = f2bf(v);
    else        out_f[i]  = v;
}

// =====================================================================
extern "C" void kernel_launch(void* const* d_in, const int* in_sizes, int n_in,
                              void* d_out, int out_size, void* d_ws, size_t ws_size,
                              hipStream_t stream) {
    const float* x  = (const float*)d_in[0];
    const int*   ei = (const int*)d_in[1];
    const float* W[3]  = {(const float*)d_in[2], (const float*)d_in[6],  (const float*)d_in[10]};
    const float* AS[3] = {(const float*)d_in[3], (const float*)d_in[7],  (const float*)d_in[11]};
    const float* AD[3] = {(const float*)d_in[4], (const float*)d_in[8],  (const float*)d_in[12]};
    const float* BI[3] = {(const float*)d_in[5], (const float*)d_in[9],  (const float*)d_in[13]};

    const int Kin[3] = {128, 256, 256};
    const int Hh[3]  = {4, 4, 1};
    const int Cc[3]  = {64, 64, 47};
    const int Ntr[3] = {256, 256, 47};
    const int nT[3]  = {16, 16, 3};     // n-tiles of 16 (layer3 padded 47 -> 48)
    const int nG[3]  = {4, 4, 1};       // wave groups across columns (NT/JT)

    char* ws = (char*)d_ws;
    size_t off = 0;
    auto alloc = [&](size_t bytes) -> void* {
        void* p = ws + off;
        off += (bytes + 255) & ~(size_t)255;
        return p;
    };
    bf16* xb = (bf16*)alloc((size_t)NNODES * 128 * sizeof(bf16));
    bf16* hb = (bf16*)alloc((size_t)NNODES * 256 * sizeof(bf16));
    bf16* Wp[3];
    Wp[0] = (bf16*)alloc((size_t)(128 / 32) * 16 * 512 * sizeof(bf16));
    Wp[1] = (bf16*)alloc((size_t)(256 / 32) * 16 * 512 * sizeof(bf16));
    Wp[2] = (bf16*)alloc((size_t)(256 / 32) * 3  * 512 * sizeof(bf16));
    float* feat  = (float*)alloc((size_t)NNODES * 256 * sizeof(float));
    float* agg   = (float*)alloc((size_t)NNODES * 256 * sizeof(float));
    float* als   = (float*)alloc((size_t)NNODES * 4 * sizeof(float));
    float* ald   = (float*)alloc((size_t)NNODES * 4 * sizeof(float));
    float* emax  = (float*)alloc((size_t)NNODES * 4 * sizeof(float));
    float* denom = (float*)alloc((size_t)NNODES * 4 * sizeof(float));
    float* esc   = (float*)alloc((size_t)ETOT * 4 * sizeof(float));
    float* ew    = (float*)alloc((size_t)ETOT * 4 * sizeof(float));
    (void)ws_size; (void)in_sizes; (void)n_in; (void)out_size;

    { int n = NNODES * 128;
      f32_to_bf16_k<<<(n + 255) / 256, 256, 0, stream>>>(x, xb, n); }

    for (int l = 0; l < 3; ++l) {
        const int K = Kin[l], H = Hh[l], C = Cc[l], Nt = Ntr[l], nTiles = nT[l];
        const int hc = H * C;  // == Nt

        { int tot = (K / 32) * nTiles * 512;
          pack_w_k<<<(tot + 255) / 256, 256, 0, stream>>>(W[l], Wp[l], K, Nt, nTiles); }

        { int waves  = (NNODES / 16) * nG[l];
          int blocks = (waves * 32 + 255) / 256;
          if (l == 0)
              gat_gemm_wmma<4, 16, 4, 256><<<blocks, 256, 0, stream>>>(xb, Wp[0], feat);
          else if (l == 1)
              gat_gemm_wmma<8, 16, 4, 256><<<blocks, 256, 0, stream>>>(hb, Wp[1], feat);
          else
              gat_gemm_wmma<8, 3, 3, 47><<<blocks, 256, 0, stream>>>(hb, Wp[2], feat);
        }

        { int tot = NNODES * H;
          att_logits_k<<<(tot + 255) / 256, 256, 0, stream>>>(feat, AS[l], AD[l],
                                                              als, ald, emax, denom, H, C); }

        hipMemsetAsync(agg, 0, (size_t)NNODES * hc * sizeof(float), stream);

        { int tot = ETOT * H;
          edge_score_max_k<<<(tot + 255) / 256, 256, 0, stream>>>(als, ald, ei, esc, emax, H);
          edge_exp_sum_k<<<(tot + 255) / 256, 256, 0, stream>>>(esc, emax, ei, ew, denom, H); }

        { int blocks = (ETOT * 32 + 255) / 256;
          aggregate_k<<<blocks, 256, 0, stream>>>(feat, ew, denom, ei, agg, H, C); }

        { int tot = NNODES * hc;
          if (l < 2)
              finalize_k<<<(tot + 255) / 256, 256, 0, stream>>>(agg, BI[l], hb, nullptr, tot, hc, 1);
          else
              finalize_k<<<(tot + 255) / 256, 256, 0, stream>>>(agg, BI[l], nullptr, (float*)d_out, tot, hc, 0);
        }
    }
}
